// HardNegativeContrastiveLoss_71665824301712
// MI455X (gfx1250) — matrix operand order repaired and checked by reading, not compiled, not run on hardware
//
#include <hip/hip_runtime.h>

#define N_TOTAL 16384
#define D_DIM   256
#define MARGIN  0.2f

// fragment-swizzled bf16 arrays: per (16-row block, 32-K block) a contiguous
// 32-lane x 16-element chunk (512 bf16 = 1KB).
#define KBLKS   (D_DIM / 32)            // 8
#define RBLKS   (N_TOTAL / 16)          // 1024
#define FRAG_ELEMS 512                  // 32 lanes * 16 bf16

typedef __attribute__((ext_vector_type(16))) __bf16 v16bf;
typedef __attribute__((ext_vector_type(8)))  float  v8f;

// Order-preserving float <-> uint mapping so unsigned atomicMax == float max.
__device__ __forceinline__ unsigned fkey(float f) {
  unsigned u = __float_as_uint(f);
  return (u & 0x80000000u) ? ~u : (u | 0x80000000u);
}
__device__ __forceinline__ float fval(unsigned k) {
  return (k & 0x80000000u) ? __uint_as_float(k & 0x7fffffffu)
                           : __uint_as_float(~k);
}

// ---------------------------------------------------------------------------
// diag[i] = <imgs[i], caps[i]> ; one wave32 per row.
// ---------------------------------------------------------------------------
__global__ void diag_kernel(const float* __restrict__ imgs,
                            const float* __restrict__ caps,
                            float* __restrict__ diag) {
  const int wave = threadIdx.x >> 5;
  const int lane = threadIdx.x & 31;
  const int row  = blockIdx.x * 8 + wave;
  const float* ip = imgs + (size_t)row * D_DIM;
  const float* cp = caps + (size_t)row * D_DIM;
  float s = 0.f;
#pragma unroll
  for (int t = 0; t < D_DIM / 32; ++t)
    s += ip[lane + 32 * t] * cp[lane + 32 * t];
#pragma unroll
  for (int off = 16; off > 0; off >>= 1)
    s += __shfl_xor(s, off, 32);
  if (lane == 0) diag[row] = s;
}

// ---------------------------------------------------------------------------
// Zero-init the 2*N uint max keys (key 0 < key(any finite float)).
// ---------------------------------------------------------------------------
__global__ void init_kernel(unsigned* __restrict__ keys) {
  keys[blockIdx.x * blockDim.x + threadIdx.x] = 0u;
}

// ---------------------------------------------------------------------------
// One-time fp32 -> bf16 hi/lo split into WMMA *A-fragment* layout.
// thread tid -> (blk, kb, lane); writes 16 hi + 16 lo bf16 at tid*16.
// A frag (16x32): lane m (=lane&15) holds row M=m, half h:
//   elems e -> K = kb*32 + 16h + e (contiguous, matches B)
// ---------------------------------------------------------------------------
__global__ void convertA_kernel(const float* __restrict__ src,
                                __bf16* __restrict__ hi,
                                __bf16* __restrict__ lo) {
  const int tid  = blockIdx.x * blockDim.x + threadIdx.x;   // 0 .. RBLKS*KBLKS*32
  const int lane = tid & 31;
  const int kb   = (tid >> 5) & (KBLKS - 1);
  const int blk  = tid >> 8;                                 // /(32*KBLKS)
  const int m = lane & 15, h = lane >> 4;
  const float* p = src + (size_t)(blk * 16 + m) * D_DIM + kb * 32;
  v16bf vh, vl;
#pragma unroll
  for (int e = 0; e < 8; ++e) {
    float f0 = p[16 * h + e];
    float f1 = p[16 * h + 8 + e];
    __bf16 h0 = (__bf16)f0, h1 = (__bf16)f1;
    vh[e] = h0;     vh[e + 8] = h1;
    vl[e] = (__bf16)(f0 - (float)h0);
    vl[e + 8] = (__bf16)(f1 - (float)h1);
  }
  *(v16bf*)(hi + (size_t)tid * 16) = vh;
  *(v16bf*)(lo + (size_t)tid * 16) = vl;
}

// ---------------------------------------------------------------------------
// Same for the *B-fragment* layout (32x16, contiguous k order): lane n (=lane&15)
// holds column N=n; elems e -> K = kb*32 + 16h + e (contiguous in memory).
// ---------------------------------------------------------------------------
__global__ void convertB_kernel(const float* __restrict__ src,
                                __bf16* __restrict__ hi,
                                __bf16* __restrict__ lo) {
  const int tid  = blockIdx.x * blockDim.x + threadIdx.x;
  const int lane = tid & 31;
  const int kb   = (tid >> 5) & (KBLKS - 1);
  const int blk  = tid >> 8;
  const int n = lane & 15, h = lane >> 4;
  const float* p = src + (size_t)(blk * 16 + n) * D_DIM + kb * 32 + 16 * h;
  v16bf vh, vl;
#pragma unroll
  for (int e = 0; e < 16; ++e) {
    float f = p[e];
    __bf16 hh = (__bf16)f;
    vh[e] = hh;
    vl[e] = (__bf16)(f - (float)hh);
  }
  *(v16bf*)(hi + (size_t)tid * 16) = vh;
  *(v16bf*)(lo + (size_t)tid * 16) = vl;
}

// ---------------------------------------------------------------------------
// Fused GEMM + diagonal fix + row/col max.
// Workgroup tile 128x128; 8 waves (2 along M x 4 along N); wave tile 64x32
// with 4x2 accumulators. bf16x3 split: acc += hi*hi + hi*lo + lo*hi.
// All fragment loads are pre-swizzled: lane-contiguous 32B (2x b128).
// ---------------------------------------------------------------------------
__launch_bounds__(256)
__global__ void gemm_kernel(const __bf16* __restrict__ a_hi,
                            const __bf16* __restrict__ a_lo,
                            const __bf16* __restrict__ b_hi,
                            const __bf16* __restrict__ b_lo,
                            const float* __restrict__ diag,
                            unsigned* __restrict__ rowkey,
                            unsigned* __restrict__ colkey) {
  const int lane  = threadIdx.x & 31;
  const int wave  = threadIdx.x >> 5;
  const int waveM = wave & 1;   // 2 waves along M
  const int waveN = wave >> 1;  // 4 waves along N
  const int m = lane & 15;
  const int h = lane >> 4;

  const int rowbase = blockIdx.y * 128 + waveM * 64;  // wave rows [rowbase,+64)
  const int colbase = blockIdx.x * 128 + waveN * 32;  // wave cols [colbase,+32)
  const int rb0 = rowbase >> 4;                       // 16-row block index
  const int cb0 = colbase >> 4;                       // 16-col block index

  v8f acc[4][2];
  {
    v8f z = {};
#pragma unroll
    for (int mt = 0; mt < 4; ++mt)
#pragma unroll
      for (int nt = 0; nt < 2; ++nt) acc[mt][nt] = z;
  }

  for (int kb = 0; kb < KBLKS; ++kb) {
    v16bf ahi[4], alo[4];
#pragma unroll
    for (int mt = 0; mt < 4; ++mt) {
      const size_t off =
          ((size_t)(rb0 + mt) * KBLKS + kb) * FRAG_ELEMS + lane * 16;
      ahi[mt] = *(const v16bf*)(a_hi + off);
      alo[mt] = *(const v16bf*)(a_lo + off);
    }
    v16bf bhi[2], blo[2];
#pragma unroll
    for (int nt = 0; nt < 2; ++nt) {
      const size_t off =
          ((size_t)(cb0 + nt) * KBLKS + kb) * FRAG_ELEMS + lane * 16;
      bhi[nt] = *(const v16bf*)(b_hi + off);
      blo[nt] = *(const v16bf*)(b_lo + off);
    }
#pragma unroll
    for (int mt = 0; mt < 4; ++mt)
#pragma unroll
      for (int nt = 0; nt < 2; ++nt) {
        acc[mt][nt] = __builtin_amdgcn_wmma_f32_16x16x32_bf16(
            false, ahi[mt], false, bhi[nt], (short)0, acc[mt][nt], false, false);
        acc[mt][nt] = __builtin_amdgcn_wmma_f32_16x16x32_bf16(
            false, ahi[mt], false, blo[nt], (short)0, acc[mt][nt], false, false);
        acc[mt][nt] = __builtin_amdgcn_wmma_f32_16x16x32_bf16(
            false, alo[mt], false, bhi[nt], (short)0, acc[mt][nt], false, false);
      }
  }

  // C/D layout: VGPR r, lanes 0-15 -> M=r, lanes 16-31 -> M=r+8; N = lane&15.
  // Diagonal fix (s[i,i] = score - 2*diag[i]) + per-row max over 32 cols.
#pragma unroll
  for (int mt = 0; mt < 4; ++mt) {
#pragma unroll
    for (int r = 0; r < 8; ++r) {
      const int grow = rowbase + 16 * mt + r + 8 * h;
      float s0 = acc[mt][0][r];
      float s1 = acc[mt][1][r];
      if (grow == colbase + m)      s0 -= 2.0f * diag[grow];
      if (grow == colbase + 16 + m) s1 -= 2.0f * diag[grow];
      acc[mt][0][r] = s0;
      acc[mt][1][r] = s1;
      float rm = fmaxf(s0, s1);
#pragma unroll
      for (int off = 8; off > 0; off >>= 1)  // xor<16: stays within lane-half
        rm = fmaxf(rm, __shfl_xor(rm, off, 32));
      if (m == 0) atomicMax(rowkey + grow, fkey(rm));
    }
  }

  // Per-column max over the wave's 64 rows.
#pragma unroll
  for (int nt = 0; nt < 2; ++nt) {
    float cm = -3.402823466e38f;
#pragma unroll
    for (int mt = 0; mt < 4; ++mt)
#pragma unroll
      for (int r = 0; r < 8; ++r) cm = fmaxf(cm, acc[mt][nt][r]);
    cm = fmaxf(cm, __shfl_xor(cm, 16, 32));  // combine the two M-halves
    if (h == 0) atomicMax(colkey + colbase + 16 * nt + m, fkey(cm));
  }
}

// ---------------------------------------------------------------------------
// loss = sum relu(rowmax + margin - diag) + sum relu(colmax + margin - diag)
// ---------------------------------------------------------------------------
__global__ void finalize_kernel(const float* __restrict__ diag,
                                const unsigned* __restrict__ rowkey,
                                const unsigned* __restrict__ colkey,
                                float* __restrict__ out) {
  __shared__ float red[256];
  float acc = 0.f;
  for (int i = threadIdx.x; i < N_TOTAL; i += 256) {
    float d = diag[i];
    acc += fmaxf(fval(rowkey[i]) + MARGIN - d, 0.f);  // neg_img (row max)
    acc += fmaxf(fval(colkey[i]) + MARGIN - d, 0.f);  // neg_cap (col max)
  }
  red[threadIdx.x] = acc;
  __syncthreads();
  for (int s = 128; s > 0; s >>= 1) {
    if (threadIdx.x < s) red[threadIdx.x] += red[threadIdx.x + s];
    __syncthreads();
  }
  if (threadIdx.x == 0) out[0] = red[0];
}

// ---------------------------------------------------------------------------
extern "C" void kernel_launch(void* const* d_in, const int* in_sizes, int n_in,
                              void* d_out, int out_size, void* d_ws,
                              size_t ws_size, hipStream_t stream) {
  (void)in_sizes; (void)n_in; (void)out_size; (void)ws_size;
  const float* imgs = (const float*)d_in[0];
  const float* caps = (const float*)d_in[1];

  char* ws = (char*)d_ws;
  float*    diag   = (float*)ws;                         // N floats
  unsigned* rowkey = (unsigned*)(ws + 4u * N_TOTAL);     // N uints
  unsigned* colkey = (unsigned*)(ws + 8u * N_TOTAL);     // N uints
  const size_t ELEMS = (size_t)N_TOTAL * D_DIM;          // 4.19M bf16 per array
  __bf16* a_hi = (__bf16*)(ws + 12u * N_TOTAL);
  __bf16* a_lo = a_hi + ELEMS;
  __bf16* b_hi = a_lo + ELEMS;
  __bf16* b_lo = b_hi + ELEMS;

  diag_kernel<<<N_TOTAL / 8, 256, 0, stream>>>(imgs, caps, diag);
  init_kernel<<<(2 * N_TOTAL) / 256, 256, 0, stream>>>(rowkey);

  const int convThreads = RBLKS * KBLKS * 32;            // 262144
  convertA_kernel<<<convThreads / 256, 256, 0, stream>>>(imgs, a_hi, a_lo);
  convertB_kernel<<<convThreads / 256, 256, 0, stream>>>(caps, b_hi, b_lo);

  dim3 grid(N_TOTAL / 128, N_TOTAL / 128);
  gemm_kernel<<<grid, 256, 0, stream>>>(a_hi, a_lo, b_hi, b_lo, diag, rowkey,
                                        colkey);

  finalize_kernel<<<1, 256, 0, stream>>>(diag, rowkey, colkey, (float*)d_out);
}
